// GraphGPS_85383949845300
// MI455X (gfx1250) — compile-verified, matrix operand order checked by reading, not run on hardware
//
#include <hip/hip_runtime.h>
#include <math.h>
#include <stdint.h>

// ---------------- CDNA5 WMMA plumbing ----------------
using bh = __bf16;
typedef __attribute__((ext_vector_type(16))) __bf16 v16bf;
typedef __attribute__((ext_vector_type(8)))  float  v8f;
typedef __attribute__((ext_vector_type(4)))  unsigned int u32x4;
typedef __attribute__((ext_vector_type(8)))  unsigned int u32x8;

__device__ __forceinline__ v16bf load_frag(const bh* p) {
  // low 8 bf16 at p, high 8 bf16 at p+16 (16-bit A/B fragment layout:
  // lane l holds row l%16; k-chunks at elem offsets g*8 and 16+g*8, g=l/16)
  union { uint4 u[2]; v16bf v; } t;
  t.u[0] = *(const uint4*)(p);
  t.u[1] = *(const uint4*)(p + 16);
  return t.v;
}

__device__ __forceinline__ v8f wmma_bf16(v16bf a, v16bf b, v8f c) {
  return __builtin_amdgcn_wmma_f32_16x16x32_bf16(false, a, false, b, (short)0, c, false, false);
}

// ---------------- Tensor Data Mover: 2-D strided tile -> LDS ----------------
// Builds a D# (groups 0/1 only, VADDR2/3 = NULL -> up to 2-D tensor) per ISA
// CDNA5 ISA and issues tensor_load_to_lds. All inputs
// must be wave-uniform. data_size = 2 bytes (bf16).
__device__ __forceinline__ void tdm_load_2d_bf16(const void* gbase, unsigned lds_off,
                                                 unsigned tile_w, unsigned tile_h,
                                                 unsigned row_stride) {
  uint64_t ga = (uint64_t)(uintptr_t)gbase;
  u32x4 g0;
  g0[0] = 1u;                                            // count=1, no gather
  g0[1] = lds_off;                                       // lds_addr (bytes)
  g0[2] = (unsigned)ga;                                  // global_addr[31:0]
  g0[3] = (unsigned)((ga >> 32) & 0x01FFFFFFu) | 0x80000000u; // addr[56:32] | type=2
  u32x8 g1;
  g1[0] = (1u << 16);                                    // data_size=1 -> 2 bytes
  g1[1] = (tile_w & 0xFFFFu) << 16;                      // tensor_dim0 lo16
  g1[2] = (tile_w >> 16) | (tile_h << 16);               // tensor_dim0 hi | tensor_dim1 lo
  g1[3] = (tile_h >> 16) | (tile_w << 16);               // tensor_dim1 hi | tile_dim0
  g1[4] = (tile_h & 0xFFFFu);                            // tile_dim1 | tile_dim2=0
  g1[5] = row_stride;                                    // tensor_dim0_stride lo32
  g1[6] = 0u;                                            // stride0 hi | stride1 lo
  g1[7] = 0u;
  asm volatile("tensor_load_to_lds %0, %1" :: "s"(g0), "s"(g1) : "memory");
}

// ---------------- small elementwise kernels ----------------
__global__ void k_fill_zero(float* p, size_t n) {
  size_t i = (size_t)blockIdx.x * blockDim.x + threadIdx.x;
  if (i < n) p[i] = 0.f;
}

__global__ void k_f2bf(const float* __restrict__ a, bh* __restrict__ o, size_t n) {
  size_t i = (size_t)blockIdx.x * blockDim.x + threadIdx.x;
  if (i < n) o[i] = (bh)a[i];
}

__global__ void k_add_convert(const float* __restrict__ a, const float* __restrict__ b,
                              bh* __restrict__ o, size_t n) {
  size_t i = (size_t)blockIdx.x * blockDim.x + threadIdx.x;
  if (i < n) o[i] = (bh)(a[i] + b[i]);
}

__global__ void k_add2(const float* __restrict__ a, const float* __restrict__ b,
                       float* __restrict__ xf, bh* __restrict__ xb, size_t n) {
  size_t i = (size_t)blockIdx.x * blockDim.x + threadIdx.x;
  if (i < n) { float v = a[i] + b[i]; xf[i] = v; xb[i] = (bh)v; }
}

// ---------------- GIN edge scatter (L2-resident f32 atomics) ----------------
__global__ void k_edge_scatter(const float* __restrict__ h, const int* __restrict__ src,
                               const int* __restrict__ dst, float* __restrict__ agg, int E) {
  size_t gid = (size_t)blockIdx.x * blockDim.x + threadIdx.x;
  size_t total = (size_t)E * 32;             // 32 chunks of 8 floats per edge
  if (gid >= total) return;
  int e = (int)(gid >> 5);
  int c = (int)(gid & 31) * 8;
  int s = src[e], d = dst[e];
  const float* hp = h + (size_t)s * 256 + c;
  float* ap = agg + (size_t)d * 256 + c;
#pragma unroll
  for (int j = 0; j < 8; ++j) atomicAdd(ap + j, hp[j]);
}

// ---------------- BatchNorm (train-mode, biased var) ----------------
__global__ void k_bn_reduce(const float* __restrict__ x, float* __restrict__ stats) {
  int col = threadIdx.x;                     // 256 cols
  size_t row0 = (size_t)blockIdx.x * 64;
  float s = 0.f, sq = 0.f;
  for (int i = 0; i < 64; ++i) {
    float v = x[(row0 + i) * 256 + col];
    s += v; sq += v * v;
  }
  atomicAdd(&stats[col], s);
  atomicAdd(&stats[256 + col], sq);
}

__global__ void k_bn_finalize(const float* __restrict__ stats, const float* __restrict__ g,
                              const float* __restrict__ b, float* __restrict__ ss, float invM) {
  int c = threadIdx.x;
  float mean = stats[c] * invM;
  float var  = stats[256 + c] * invM - mean * mean;
  float sc   = g[c] * rsqrtf(var + 1e-5f);
  ss[c] = sc;
  ss[256 + c] = b[c] - mean * sc;
}

__global__ void k_bn_apply(const float* __restrict__ x, const float* __restrict__ ss,
                           float* __restrict__ y, size_t n) {
  size_t i = (size_t)blockIdx.x * blockDim.x + threadIdx.x;
  if (i < n) {
    int c = (int)(i & 255);
    y[i] = x[i] * ss[c] + ss[256 + c];
  }
}

// ---------------- generic WMMA linear: out = act(X @ W^T + bias [+ resid]) ----------------
// X: [M,K] bf16, W: [N,K] bf16 row-major, out: [M,N]. 128 threads / 4 waves,
// block tile 64 rows x 32 cols; W tile (32 x K) staged into LDS by the TDM.
template<bool RELU, bool RESID, bool OUTBF>
__global__ __launch_bounds__(128)
void k_gemm(const bh* __restrict__ X, const bh* __restrict__ W,
            const float* __restrict__ bias, const float* __restrict__ resid,
            void* __restrict__ outp, int N, int K) {
  __shared__ __align__(16) bh Wl[32 * 512];
  const int tid = threadIdx.x;
  const int wave = tid >> 5, lane = tid & 31;
  const int m0 = blockIdx.x * 64;
  const int n0 = blockIdx.y * 32;
  if (wave == 0) {
    tdm_load_2d_bf16(W + (size_t)n0 * K, (unsigned)(uintptr_t)Wl,
                     (unsigned)K, 32u, (unsigned)K);
    __builtin_amdgcn_s_wait_tensorcnt(0);
  }
  __syncthreads();
  const int g = lane >> 4, lm = lane & 15;
  const size_t arow = (size_t)(m0 + wave * 16 + lm);
  const bh* ab0 = X + arow * K + g * 8;
  const bh* bb0 = &Wl[lm * K + g * 8];
  const bh* bb1 = &Wl[(16 + lm) * K + g * 8];
  const v8f vz = {0.f,0.f,0.f,0.f,0.f,0.f,0.f,0.f};
  v8f acc[2]; acc[0] = vz; acc[1] = vz;
  for (int k0 = 0; k0 < K; k0 += 32) {
    v16bf A  = load_frag(ab0 + k0);
    v16bf B0 = load_frag(bb0 + k0);
    v16bf B1 = load_frag(bb1 + k0);
    acc[0] = wmma_bf16(A, B0, acc[0]);
    acc[1] = wmma_bf16(A, B1, acc[1]);
  }
#pragma unroll
  for (int nt = 0; nt < 2; ++nt) {
    const int col = n0 + nt * 16 + lm;
    const float bv = bias[col];
#pragma unroll
    for (int r = 0; r < 8; ++r) {
      const size_t row = (size_t)m0 + wave * 16 + r + 8 * g;  // C layout: M = r + 8*(l/16)
      float v = acc[nt][r] + bv;
      if (RESID) v += resid[row * (size_t)N + col];
      if (RELU)  v = fmaxf(v, 0.f);
      if (OUTBF) ((bh*)outp)[row * (size_t)N + col] = (bh)v;
      else       ((float*)outp)[row * (size_t)N + col] = v;
    }
  }
}

// ---------------- per-(graph,head) attention: N=128, HD=32 ----------------
__global__ __launch_bounds__(256)
void k_attn(const bh* __restrict__ qkv, bh* __restrict__ ctx) {
  __shared__ __align__(16) bh Kl[128 * 32];     // K tile, row-major (TDM-loaded)
  __shared__ __align__(16) bh VT[32 * 128];     // V transposed
  __shared__ __align__(16) bh Pl[8 * 16 * 128]; // probs per wave
  const int blk = blockIdx.x;
  const int b = blk >> 3, hI = blk & 7;
  const int rowBase = b * 128;
  const int tid = threadIdx.x;
  const int wave = tid >> 5, lane = tid & 31;
  // K tile: 128 rows x 32 cols, row stride 768 elems -> one TDM descriptor.
  if (wave == 0) {
    tdm_load_2d_bf16(qkv + (size_t)rowBase * 768 + 256 + hI * 32,
                     (unsigned)(uintptr_t)Kl, 32u, 128u, 768u);
  }
  // V transposed by all threads while the DMA runs
  for (int i = tid; i < 4096; i += 256) {
    int r = i >> 5, c = i & 31;
    VT[c * 128 + r] = qkv[(size_t)(rowBase + r) * 768 + 512 + hI * 32 + c];
  }
  if (wave == 0) __builtin_amdgcn_s_wait_tensorcnt(0);
  __syncthreads();
  const int g = lane >> 4, lm = lane & 15;
  const v8f vz = {0.f,0.f,0.f,0.f,0.f,0.f,0.f,0.f};
  // scores: wave handles query rows [wave*16, wave*16+16), full 128 key cols
  const bh* qb = qkv + (size_t)(rowBase + wave * 16 + lm) * 768 + hI * 32 + g * 8;
  v16bf Af = load_frag(qb);                     // 16x32 Q fragment (HD=32 in one step)
  v8f S[8];
#pragma unroll
  for (int t = 0; t < 8; ++t) {
    v16bf Bf = load_frag(&Kl[(t * 16 + lm) * 32 + g * 8]);
    S[t] = wmma_bf16(Af, Bf, vz);
  }
  const float scale = 0.17677669529663689f;     // 1/sqrt(32)
#pragma unroll
  for (int r = 0; r < 8; ++r) {                 // softmax per row, 16-lane shfl reduce
    float mx = -3.0e38f;
#pragma unroll
    for (int t = 0; t < 8; ++t) { S[t][r] *= scale; mx = fmaxf(mx, S[t][r]); }
#pragma unroll
    for (int m = 8; m >= 1; m >>= 1) mx = fmaxf(mx, __shfl_xor(mx, m, 32));
    float sum = 0.f;
#pragma unroll
    for (int t = 0; t < 8; ++t) { float e = __expf(S[t][r] - mx); S[t][r] = e; sum += e; }
#pragma unroll
    for (int m = 8; m >= 1; m >>= 1) sum += __shfl_xor(sum, m, 32);
    float inv = 1.0f / sum;
#pragma unroll
    for (int t = 0; t < 8; ++t)
      Pl[wave * 2048 + (r + 8 * g) * 128 + t * 16 + lm] = (bh)(S[t][r] * inv);
  }
  __syncthreads();
#pragma unroll
  for (int ct = 0; ct < 2; ++ct) {              // O = P[16,128] @ V[128,32]
    v8f O = vz;
#pragma unroll
    for (int ks = 0; ks < 4; ++ks) {
      v16bf Ap = load_frag(&Pl[wave * 2048 + lm * 128 + ks * 32 + g * 8]);
      v16bf Bv = load_frag(&VT[(ct * 16 + lm) * 128 + ks * 32 + g * 8]);
      O = wmma_bf16(Ap, Bv, O);
    }
#pragma unroll
    for (int r = 0; r < 8; ++r) {
      size_t row = (size_t)rowBase + wave * 16 + r + 8 * g;
      ctx[row * 256 + hI * 32 + ct * 16 + lm] = (bh)O[r];
    }
  }
}

// ---------------- host orchestration ----------------
static inline dim3 grid1(size_t n, int b) { return dim3((unsigned)((n + b - 1) / b)); }

extern "C" void kernel_launch(void* const* d_in, const int* in_sizes, int n_in,
                              void* d_out, int out_size, void* d_ws, size_t ws_size,
                              hipStream_t stream) {
  const float* h    = (const float*)d_in[0];
  const int*   esrc = (const int*)d_in[1];
  const int*   edst = (const int*)d_in[2];
  const float* gw1 = (const float*)d_in[3];  const float* gb1 = (const float*)d_in[4];
  const float* gw2 = (const float*)d_in[5];  const float* gb2 = (const float*)d_in[6];
  const float* aiw = (const float*)d_in[7];  const float* aib = (const float*)d_in[8];
  const float* aow = (const float*)d_in[9];  const float* aob = (const float*)d_in[10];
  const float* blg = (const float*)d_in[11]; const float* blb = (const float*)d_in[12];
  const float* bag = (const float*)d_in[13]; const float* bab = (const float*)d_in[14];
  const float* bog = (const float*)d_in[15]; const float* bob = (const float*)d_in[16];
  const float* f1w = (const float*)d_in[17]; const float* f1b = (const float*)d_in[18];
  const float* f2w = (const float*)d_in[19]; const float* f2b = (const float*)d_in[20];

  const int M = 65536;                 // B*N nodes
  const size_t NHsz = (size_t)M * 256; // node-feature element count
  const int E = in_sizes[1];

  // workspace carve (with dead-buffer aliasing)
  char* ws = (char*)d_ws; size_t off = 0;
  auto carve = [&](size_t bytes) -> char* {
    char* p = ws + off; off += (bytes + 255) & ~(size_t)255; return p;
  };
  bh* wb1 = (bh*)carve(256 * 256 * 2);
  bh* wb2 = (bh*)carve(256 * 256 * 2);
  bh* wbi = (bh*)carve(768 * 256 * 2);
  bh* wbo = (bh*)carve(256 * 256 * 2);
  bh* wf1 = (bh*)carve(512 * 256 * 2);
  bh* wf2 = (bh*)carve(256 * 512 * 2);
  float* stats = (float*)carve(512 * 4);
  float* ss    = (float*)carve(512 * 4);
  bh* hb       = (bh*)carve(NHsz * 2);
  float* agg   = (float*)carve(NHsz * 4);  float* bufA = agg;    // agg dead before bufA live
  bh* zb       = (bh*)carve(NHsz * 2);     bh* ctx = zb;         // zb dead before ctx live
  bh* g1b      = (bh*)carve(NHsz * 2);     bh* xb  = g1b;        // g1 dead before xb live
  float* bufB  = (float*)carve(NHsz * 4);
  bh* qkvb     = (bh*)carve((size_t)M * 768 * 2);  bh* ffnh = qkvb; // qkv dead before ffn live
  (void)ws_size; (void)n_in; (void)out_size;

  // weight + input conversions to bf16
  k_f2bf<<<grid1(256*256,256),256,0,stream>>>(gw1, wb1, 256*256);
  k_f2bf<<<grid1(256*256,256),256,0,stream>>>(gw2, wb2, 256*256);
  k_f2bf<<<grid1(768*256,256),256,0,stream>>>(aiw, wbi, 768*256);
  k_f2bf<<<grid1(256*256,256),256,0,stream>>>(aow, wbo, 256*256);
  k_f2bf<<<grid1(512*256,256),256,0,stream>>>(f1w, wf1, 512*256);
  k_f2bf<<<grid1(256*512,256),256,0,stream>>>(f2w, wf2, 256*512);
  k_f2bf<<<grid1(NHsz,256),256,0,stream>>>(h, hb, NHsz);

  // --- GIN branch ---
  k_fill_zero<<<grid1(NHsz,256),256,0,stream>>>(agg, NHsz);
  k_edge_scatter<<<grid1((size_t)E*32,256),256,0,stream>>>(h, esrc, edst, agg, E);
  k_add_convert<<<grid1(NHsz,256),256,0,stream>>>(h, agg, zb, NHsz);
  k_gemm<true ,false,true ><<<dim3(1024, 8),128,0,stream>>>(zb , wb1, gb1, nullptr, g1b , 256, 256);
  k_gemm<false,true ,false><<<dim3(1024, 8),128,0,stream>>>(g1b, wb2, gb2, h      , bufA, 256, 256);
  k_fill_zero<<<grid1(512,256),256,0,stream>>>(stats, 512);
  k_bn_reduce<<<dim3(1024),256,0,stream>>>(bufA, stats);
  k_bn_finalize<<<1,256,0,stream>>>(stats, blg, blb, ss, 1.0f / M);
  k_bn_apply<<<grid1(NHsz,256),256,0,stream>>>(bufA, ss, bufA, NHsz);

  // --- attention branch ---
  k_gemm<false,false,true ><<<dim3(1024,24),128,0,stream>>>(hb, wbi, aib, nullptr, qkvb, 768, 256);
  k_attn<<<dim3(4096),256,0,stream>>>(qkvb, ctx);
  k_gemm<false,true ,false><<<dim3(1024, 8),128,0,stream>>>(ctx, wbo, aob, h, bufB, 256, 256);
  k_fill_zero<<<grid1(512,256),256,0,stream>>>(stats, 512);
  k_bn_reduce<<<dim3(1024),256,0,stream>>>(bufB, stats);
  k_bn_finalize<<<1,256,0,stream>>>(stats, bag, bab, ss, 1.0f / M);
  k_bn_apply<<<grid1(NHsz,256),256,0,stream>>>(bufB, ss, bufB, NHsz);

  // --- combine + FFN + out BN ---
  k_add2<<<grid1(NHsz,256),256,0,stream>>>(bufA, bufB, bufA, xb, NHsz);
  k_gemm<true ,false,true ><<<dim3(1024,16),128,0,stream>>>(xb  , wf1, f1b, nullptr, ffnh, 512, 256);
  k_gemm<false,true ,false><<<dim3(1024, 8),128,0,stream>>>(ffnh, wf2, f2b, bufA   , bufB, 256, 512);
  k_fill_zero<<<grid1(512,256),256,0,stream>>>(stats, 512);
  k_bn_reduce<<<dim3(1024),256,0,stream>>>(bufB, stats);
  k_bn_finalize<<<1,256,0,stream>>>(stats, bog, bob, ss, 1.0f / M);
  k_bn_apply<<<grid1(NHsz,256),256,0,stream>>>(bufB, ss, (float*)d_out, NHsz);
}
